// _BRITS_25297357373579
// MI455X (gfx1250) — matrix operand order repaired and checked by reading, not compiled, not run on hardware
//
#include <hip/hip_runtime.h>
#include <hip/hip_bf16.h>

// ---------------- problem constants ----------------
constexpr int Bc = 512;     // batch
constexpr int Tc = 128;     // time steps
constexpr int Fc = 64;      // features
constexpr int Hc = 128;     // hidden
constexpr size_t BTF = (size_t)Bc * Tc * Fc;          // 4,194,304
constexpr size_t OUT_FREC = BTF;
constexpr size_t OUT_BREC = 2 * BTF;
constexpr size_t OUT_FH   = 3 * BTF;                  // [B,H] fwd final h
constexpr size_t OUT_CONS = 3 * BTF + 2 * (size_t)Bc * Hc;
constexpr size_t OUT_RECL = OUT_CONS + 1;

// fp16 weight workspace layout (offsets in halves, per-direction)
constexpr size_t WDIR  = 163840;
constexpr size_t OW_TH = 0;       // Wth  [128][64]
constexpr size_t OW_HR = 8192;    // Whr  [64][128]
constexpr size_t OW_FR = 16384;   // Wfr  [64][64] (off-diag masked)
constexpr size_t OW_CW = 20480;   // Wcw  [64][128]
constexpr size_t OW_IH = 28672;   // W_ih [512][128]
constexpr size_t OW_HH = 94208;   // W_hh [512][128]
constexpr size_t WBYTES = 2 * WDIR * 2;               // 655,360 B

// fp32 workspace (floats, after WBYTES): per-dir 960 bias floats, then loss accum
constexpr int BO_GS = 0;    // b_ih+b_hh   [512]
constexpr int BO_TH = 512;  // bth         [128]
constexpr int BO_TX = 640;  // btx         [64]
constexpr int BO_HR = 704;  // bhr         [64]
constexpr int BO_FR = 768;  // bfr         [64]
constexpr int BO_CW = 832;  // bcw         [64]
constexpr int BO_DG = 896;  // diag(Wtx)   [64]
constexpr int LNUM  = 1920; // [2][128] loss numerators
constexpr int LDEN  = 2176; // [2][128] mask sums

typedef _Float16 half8 __attribute__((ext_vector_type(8)));
typedef _Float16 v16h  __attribute__((ext_vector_type(16)));
typedef float    v8f   __attribute__((ext_vector_type(8)));

struct InPtrs { const float* p[34]; };

// ---------------- WMMA fragment helpers (contiguous k order, same in A and B) ----------------
// A (16x32 f16): lane L -> row M=L&15; K run kb..kb+15, kb=(L>>4)*16
__device__ inline v16h load_a_frag(const _Float16* a, int lda, int k0, int lane) {
  int m = lane & 15, kb = (lane >> 4) << 4;
  half8 lo = *(const half8*)(a + m * lda + k0 + kb);
  half8 hi = *(const half8*)(a + m * lda + k0 + kb + 8);
  return __builtin_shufflevector(lo, hi, 0,1,2,3,4,5,6,7,8,9,10,11,12,13,14,15);
}
// B (32x16 f16) from row-major W[N][K]: lane L -> col N=n0+(L&15); K run kb..kb+15, kb=(L>>4)*16
__device__ inline v16h load_b_frag(const _Float16* w, int ldk, int n0, int k0, int lane) {
  int n = n0 + (lane & 15), kb = (lane >> 4) << 4;
  const half8* p = (const half8*)(w + (size_t)n * ldk + k0 + kb);
  half8 lo = p[0], hi = p[1];
  return __builtin_shufflevector(lo, hi, 0,1,2,3,4,5,6,7,8,9,10,11,12,13,14,15);
}
__device__ inline v8f wmma16(v16h a, v16h b, v8f c) {
  return __builtin_amdgcn_wmma_f32_16x16x32_f16(false, a, false, b, (short)0, c, false, false);
}
template <int K>
__device__ inline v8f gemm_tile(v8f acc, const _Float16* a, int lda,
                                const _Float16* w, int ldk, int n0, int lane) {
#pragma unroll
  for (int k0 = 0; k0 < K; k0 += 32)
    acc = wmma16(load_a_frag(a, lda, k0, lane), load_b_frag(w, ldk, n0, k0, lane), acc);
  return acc;
}
__device__ inline float sigm(float x) { return 1.f / (1.f + __expf(-x)); }

// async global -> LDS copy (16B), tracked by ASYNCcnt (ISA 10 / §15.18, opcode 98)
__device__ inline void async_g2l_b128(void* lds_ptr, const void* gptr) {
  unsigned laddr = (unsigned)(uintptr_t)lds_ptr;   // LDS aperture: low 32 bits = LDS offset
  asm volatile("global_load_async_to_lds_b128 %0, %1, off"
               :: "v"(laddr), "v"(gptr) : "memory");
}
__device__ inline void wait_asynccnt0() {
  asm volatile("s_wait_asynccnt 0x0" ::: "memory");
}

// ---------------- kernel 0: zero loss accumulators ----------------
__global__ void brits_zero(float* f32ws, float* out) {
  int i = blockIdx.x * blockDim.x + threadIdx.x;
  if (i < 512) f32ws[LNUM + i] = 0.f;
  if (i == 0) out[OUT_CONS] = 0.f;
}

// ---------------- kernel 1: weights -> fp16, pack biases ----------------
__global__ void brits_prep(InPtrs P, _Float16* w16, float* f32ws) {
  const int PERDIR = 159744 + 960;
  int idx = blockIdx.x * blockDim.x + threadIdx.x;
  if (idx >= 2 * PERDIR) return;
  int dir = idx / PERDIR, r = idx % PERDIR;
  const float* const* pp = P.p + 6 + dir * 14;  // params_f / params_b
  _Float16* wd = w16 + (size_t)dir * WDIR;
  float* bd = f32ws + dir * 960;
  if (r < 159744) {
    if (r < 8192)        wd[OW_TH + r] = (_Float16)pp[4][r];                       // Wth
    else if (r < 16384)  wd[OW_HR + (r - 8192)] = (_Float16)pp[8][r - 8192];       // Whr
    else if (r < 20480) { int q = r - 16384; int i2 = q >> 6, j2 = q & 63;         // Wfr off-diag
                          wd[OW_FR + q] = (_Float16)((i2 == j2) ? 0.f : pp[10][q]); }
    else if (r < 28672)  wd[OW_CW + (r - 20480)] = (_Float16)pp[12][r - 20480];    // Wcw
    else if (r < 94208)  wd[OW_IH + (r - 28672)] = (_Float16)pp[0][r - 28672];     // W_ih
    else                 wd[OW_HH + (r - 94208)] = (_Float16)pp[1][r - 94208];     // W_hh
  } else {
    int rb = r - 159744;
    if (rb < 512)       bd[rb] = pp[2][rb] + pp[3][rb];       // b_ih + b_hh
    else if (rb < 640)  bd[rb] = pp[5][rb - 512];             // bth
    else if (rb < 704)  bd[rb] = pp[7][rb - 640];             // btx
    else if (rb < 768)  bd[rb] = pp[9][rb - 704];             // bhr
    else if (rb < 832)  bd[rb] = pp[11][rb - 768];            // bfr
    else if (rb < 896)  bd[rb] = pp[13][rb - 832];            // bcw
    else { int k = rb - 896; bd[rb] = pp[6][k * 65]; }        // diag(Wtx)
  }
}

// ---------------- kernel 2: RITS scan (one 16-row batch tile per block) ----------------
__global__ __launch_bounds__(128) void brits_rits(InPtrs P, const _Float16* __restrict__ w16,
                                                  float* __restrict__ f32ws,
                                                  float* __restrict__ out) {
  const int dir = blockIdx.y;
  const int b0  = blockIdx.x * 16;
  const int tid = threadIdx.x, lane = tid & 31, wv = tid >> 5;

  const float* __restrict__ X  = dir ? P.p[3] : P.p[0];
  const float* __restrict__ Mm = dir ? P.p[4] : P.p[1];
  const float* __restrict__ Dd = dir ? P.p[5] : P.p[2];
  const _Float16* __restrict__ Wg = w16 + (size_t)dir * WDIR;
  const float* __restrict__ bias = f32ws + dir * 960;
  float* lnum = f32ws + LNUM + dir * Tc;
  float* lden = f32ws + LDEN + dir * Tc;
  const size_t orecbase = dir ? OUT_BREC : OUT_FREC;

  __shared__ __align__(16) float h32[16][128];
  __shared__ __align__(16) float c32[16][128];
  __shared__ __align__(16) float x32[16][64];
  __shared__ __align__(16) float m32[16][64];
  __shared__ __align__(16) float d32[16][64];
  __shared__ __align__(16) _Float16 d16[16][64];
  __shared__ __align__(16) _Float16 h16[16][128];
  __shared__ __align__(16) _Float16 xc16[16][64];
  __shared__ __align__(16) _Float16 gm16[16][128];  // [gamma_x | m]
  __shared__ __align__(16) _Float16 cm16[16][128];  // [c_c | m]

  for (int idx = tid; idx < 16 * 128; idx += 128) {
    h32[idx >> 7][idx & 127] = 0.f;
    c32[idx >> 7][idx & 127] = 0.f;
  }
  __syncthreads();

  const int rowbase = (lane >> 4) << 3;  // C/D layout: rows rowbase..rowbase+7
  for (int t = 0; t < Tc; ++t) {
    float myNum = 0.f, myDen = 0.f;
    const int trec = dir ? (Tc - 1 - t) : t;

    // P0: async DMA of x, m, d tile straight into LDS (no VGPR round trip)
    for (int q = tid; q < 256; q += 128) {
      int i = q >> 4, c4 = (q & 15) << 2;
      size_t src = (size_t)(b0 + i) * Tc * Fc + (size_t)t * Fc + c4;
      async_g2l_b128(&x32[i][c4], X + src);
      async_g2l_b128(&m32[i][c4], Mm + src);
      async_g2l_b128(&d32[i][c4], Dd + src);
      if (t + 1 < Tc) __builtin_prefetch(X + src + Fc, 0, 1);   // global_prefetch_b8
    }
    wait_asynccnt0();
    __syncthreads();

    // P0b: stage d16 (f16 A operand for gamma_h GEMM)
    for (int idx = tid; idx < 16 * 64; idx += 128) {
      int i = idx >> 6, j = idx & 63;
      d16[i][j] = (_Float16)d32[i][j];
    }
    __syncthreads();

    // P1: gamma_h = exp(-relu(d @ Wth^T + bth)); h *= gamma_h   (N=128, K=64)
    for (int it = 0; it < 2; ++it) {
      int n0 = ((wv << 1) + it) << 4;
      v8f acc = {};
      acc = gemm_tile<64>(acc, &d16[0][0], 64, Wg + OW_TH, 64, n0, lane);
      int col = n0 + (lane & 15);
      float bv = bias[BO_TH + col];
#pragma unroll
      for (int r = 0; r < 8; ++r)
        h32[rowbase + r][col] *= __expf(-fmaxf(acc[r] + bv, 0.f));
    }
    __syncthreads();

    // P2: stage h16 and gm16 = [gamma_x | m]
    for (int idx = tid; idx < 16 * 128; idx += 128) {
      int i = idx >> 7, j = idx & 127;
      h16[i][j] = (_Float16)h32[i][j];
      float v = (j < 64)
        ? __expf(-fmaxf(d32[i][j] * bias[BO_DG + j] + bias[BO_TX + j], 0.f))
        : m32[i][j - 64];
      gm16[i][j] = (_Float16)v;
    }
    __syncthreads();

    // P3: x_h = h @ Whr^T + bhr  (per-wave 16-col tile, result kept in regs)
    const int n0 = wv << 4;
    const int col = n0 + (lane & 15);
    v8f xh = {};
    xh = gemm_tile<128>(xh, &h16[0][0], 128, Wg + OW_HR, 128, n0, lane);
    float xs[8], ms[8];
    {
      float bv = bias[BO_HR + col];
#pragma unroll
      for (int r = 0; r < 8; ++r) {
        int row = rowbase + r;
        xh[r] += bv;
        xs[r] = x32[row][col]; ms[r] = m32[row][col];
        myNum += fabsf(xh[r] - xs[r]) * ms[r];
        xc16[row][col] = (_Float16)(ms[r] * xs[r] + (1.f - ms[r]) * xh[r]);
      }
    }
    __syncthreads();

    // P4: z_h, alpha, c_h; write recon + stage cm16
    {
      v8f zh = {}, aa = {};
      zh = gemm_tile<64>(zh, &xc16[0][0], 64, Wg + OW_FR, 64, n0, lane);
      aa = gemm_tile<128>(aa, &gm16[0][0], 128, Wg + OW_CW, 128, n0, lane);
      float bfrv = bias[BO_FR + col], bcwv = bias[BO_CW + col];
#pragma unroll
      for (int r = 0; r < 8; ++r) {
        int row = rowbase + r;
        float z  = zh[r] + bfrv;
        myNum += fabsf(z - xs[r]) * ms[r];
        float al = sigm(aa[r] + bcwv);
        float ch = al * z + (1.f - al) * xh[r];
        myNum += fabsf(ch - xs[r]) * ms[r];
        myDen += ms[r];
        out[orecbase + (size_t)(b0 + row) * Tc * Fc + (size_t)trec * Fc + col] = ch;
        cm16[row][col] = (_Float16)(ms[r] * xs[r] + (1.f - ms[r]) * ch);
      }
      for (int idx = tid; idx < 16 * 64; idx += 128) {
        int i = idx >> 6, j = idx & 63;
        cm16[i][64 + j] = (_Float16)m32[i][j];
      }
    }
    __syncthreads();

    // P5: LSTM gates = [c_c|m] @ W_ih^T + h @ W_hh^T (+ fused biases), cell update
    for (int jt = wv; jt < 8; jt += 4) {
      int j0 = jt << 4;
      int j  = j0 + (lane & 15);
      v8f aI = {}, aF = {}, aG = {}, aO = {};
      aI = gemm_tile<128>(aI, &cm16[0][0], 128, Wg + OW_IH, 128, 0 * Hc + j0, lane);
      aI = gemm_tile<128>(aI, &h16[0][0], 128, Wg + OW_HH, 128, 0 * Hc + j0, lane);
      aF = gemm_tile<128>(aF, &cm16[0][0], 128, Wg + OW_IH, 128, 1 * Hc + j0, lane);
      aF = gemm_tile<128>(aF, &h16[0][0], 128, Wg + OW_HH, 128, 1 * Hc + j0, lane);
      aG = gemm_tile<128>(aG, &cm16[0][0], 128, Wg + OW_IH, 128, 2 * Hc + j0, lane);
      aG = gemm_tile<128>(aG, &h16[0][0], 128, Wg + OW_HH, 128, 2 * Hc + j0, lane);
      aO = gemm_tile<128>(aO, &cm16[0][0], 128, Wg + OW_IH, 128, 3 * Hc + j0, lane);
      aO = gemm_tile<128>(aO, &h16[0][0], 128, Wg + OW_HH, 128, 3 * Hc + j0, lane);
      float bI = bias[BO_GS + j], bF = bias[BO_GS + Hc + j];
      float bG = bias[BO_GS + 2 * Hc + j], bO = bias[BO_GS + 3 * Hc + j];
#pragma unroll
      for (int r = 0; r < 8; ++r) {
        int row = rowbase + r;
        float ig = sigm(aI[r] + bI), fg = sigm(aF[r] + bF);
        float gg = tanhf(aG[r] + bG), og = sigm(aO[r] + bO);
        float cN = fg * c32[row][j] + ig * gg;
        c32[row][j] = cN;
        h32[row][j] = og * tanhf(cN);
      }
    }

    // P6: per-step batch-wide MAE partials
    for (int off = 16; off > 0; off >>= 1) {
      myNum += __shfl_xor(myNum, off, 32);
      myDen += __shfl_xor(myDen, off, 32);
    }
    if (lane == 0) { atomicAdd(&lnum[t], myNum); atomicAdd(&lden[t], myDen); }
    __syncthreads();
  }

  // final hidden state -> out
  size_t hbase = OUT_FH + (size_t)dir * Bc * Hc;
  for (int idx = tid; idx < 16 * 128; idx += 128) {
    int i = idx >> 7, j = idx & 127;
    out[hbase + (size_t)(b0 + i) * Hc + j] = h32[i][j];
  }
}

// ---------------- kernel 3: imputed data + consistency partials ----------------
__global__ __launch_bounds__(256) void brits_finalize_elem(const float* __restrict__ X,
                                                           const float* __restrict__ Mm,
                                                           float* __restrict__ out) {
  size_t stride = (size_t)gridDim.x * blockDim.x;
  float acc = 0.f;
  for (size_t i = (size_t)blockIdx.x * blockDim.x + threadIdx.x; i < BTF; i += stride) {
    float x = X[i], m = Mm[i];
    float fr = out[OUT_FREC + i], br = out[OUT_BREC + i];
    out[i] = m * x + (1.f - m) * 0.5f * (fr + br);
    acc += fabsf((1.f - m) * (fr - br));
  }
  for (int off = 16; off > 0; off >>= 1) acc += __shfl_xor(acc, off, 32);
  if ((threadIdx.x & 31) == 0) atomicAdd(&out[OUT_CONS], acc);
}

// ---------------- kernel 4: scalar losses ----------------
__global__ void brits_finalize_scalar(const float* __restrict__ f32ws, float* __restrict__ out) {
  if (blockIdx.x == 0 && threadIdx.x == 0) {
    float rec = 0.f;
    for (int dir = 0; dir < 2; ++dir) {
      const float* ln = f32ws + LNUM + dir * Tc;
      const float* ld = f32ws + LDEN + dir * Tc;
      float s = 0.f;
      for (int t = 0; t < Tc; ++t) s += ln[t] / (ld[t] + 1e-12f);
      rec += s / (Tc * 3.f);
    }
    out[OUT_RECL] = rec;
    out[OUT_CONS] = out[OUT_CONS] * 0.1f / (float)BTF;
  }
}

extern "C" void kernel_launch(void* const* d_in, const int* in_sizes, int n_in,
                              void* d_out, int out_size, void* d_ws, size_t ws_size,
                              hipStream_t stream) {
  (void)in_sizes; (void)out_size; (void)ws_size;
  InPtrs P;
  for (int i = 0; i < 34; ++i) P.p[i] = (const float*)d_in[i < n_in ? i : 0];
  _Float16* w16 = (_Float16*)d_ws;
  float* f32ws = (float*)((char*)d_ws + WBYTES);
  float* out = (float*)d_out;

  brits_zero<<<2, 256, 0, stream>>>(f32ws, out);
  const int prepN = 2 * (159744 + 960);
  brits_prep<<<(prepN + 255) / 256, 256, 0, stream>>>(P, w16, f32ws);
  dim3 grid(Bc / 16, 2);
  brits_rits<<<grid, 128, 0, stream>>>(P, w16, f32ws, out);
  brits_finalize_elem<<<4096, 256, 0, stream>>>((const float*)d_in[0], (const float*)d_in[1], out);
  brits_finalize_scalar<<<1, 32, 0, stream>>>(f32ws, out);
}
